// JumpODEEncoder_17970143166667
// MI455X (gfx1250) — compile-verified, matrix operand order checked by reading, not run on hardware
//
#include <hip/hip_runtime.h>
#include <hip/hip_bf16.h>
#include <math.h>

// ---------------------------------------------------------------------------
// JumpODE encoder for MI455X (gfx1250, wave32, WMMA).
//  Phase 1: sequential RK4-ODE + GRU scan, one workgroup (16 waves), state in
//           LDS, all GEMMs via v_wmma_f32_16x16x32_f16 (M=16 == batch).
//           Weight B-fragments are re-read from packed global memory every step.
//  Phase 2: 2-layer pre-LN transformer: WMMA GEMMs with pre-packed f16 B
//           fragments (+ global_prefetch on the weight stream),
//           flash-attention with WMMA QK^T and PV.
// ---------------------------------------------------------------------------

typedef __attribute__((ext_vector_type(16))) _Float16 v16h;
typedef __attribute__((ext_vector_type(8)))  float    v8f;

#define DEV __device__ __forceinline__

constexpr int Bsz   = 16;
constexpr int Tlen  = 1024;
constexpr int INF_  = 64;     // input feature dim
constexpr int HID   = 256;
constexpr int HEADS = 4;
constexpr int DH    = 64;
constexpr int LAYERS= 2;
constexpr int FF    = 1024;
constexpr int ROWS  = Bsz * Tlen;   // 16384

DEV v8f wmma16(v16h a, v16h b, v8f c) {
  return __builtin_amdgcn_wmma_f32_16x16x32_f16(false, a, false, b, (short)0, c, false, false);
}

// --- fragment loaders ------------------------------------------------------
// A-matrix 16x32 f16 layout (ISA 7.12.2): lane = M%16 (both halves), K pattern:
// lanes 0-15: V0..3 -> K 0..7, V4..7 -> K 16..23 ; lanes 16-31: K 8..15, 24..31
DEV int kpair_of(int lane, int v2) {
  int kb = (lane >> 4) << 3;                       // 0 or 8
  return (v2 < 4) ? (kb + 2 * v2) : (16 + kb + 2 * (v2 - 4));
}

// load A fragment from row-major f16 matrix tile (16 rows x 32 cols)
DEV v16h load_a_frag(const _Float16* base, int stride, int lane) {
  union { v16h v; unsigned u[8]; } r;
  int m = lane & 15;
  const _Float16* rowp = base + (size_t)m * stride;
#pragma unroll
  for (int v2 = 0; v2 < 8; ++v2) {
    int kv = kpair_of(lane, v2);
    r.u[v2] = *(const unsigned*)(rowp + kv);       // two consecutive halves
  }
  return r.v;
}

// load B fragment whose K runs over matrix *rows* (for K^T-free attention PV):
// value(k,n) = base[k*stride + n], n = lane%16
DEV v16h load_b_rows(const _Float16* base, int stride, int lane) {
  union { v16h v; _Float16 h[16]; } r;
  int n = lane & 15;
#pragma unroll
  for (int v2 = 0; v2 < 8; ++v2) {
    int kv = kpair_of(lane, v2);
    r.h[2 * v2]     = base[(size_t)kv * stride + n];
    r.h[2 * v2 + 1] = base[(size_t)(kv + 1) * stride + n];
  }
  return r.v;
}

DEV v8f vzero8() {
  v8f c;
#pragma unroll
  for (int j = 0; j < 8; ++j) c[j] = 0.f;
  return c;
}

// packed-B GEMM inner loop: acc += A(a_src tile) * Bpacked(ntile)
DEV void mm_tile(const _Float16* a_src, int astride, const _Float16* Wp,
                 int KT, int ntile, int lane, v8f& acc) {
  for (int kt = 0; kt < KT; ++kt) {
    v16h a = load_a_frag(a_src + kt * 32, astride, lane);
    v16h b = *(const v16h*)(Wp + (((size_t)ntile * KT + kt) * 32 + lane) * 16);
    acc = wmma16(a, b, acc);
  }
}

// fast sigmoid: exp + hardware reciprocal (v_rcp_f32), no IEEE divide
DEV float sigmoidf_(float x) {
  return __builtin_amdgcn_rcpf(1.f + __expf(-x));
}

// branch-free tanh: two transcendentals, no IEEE divide expansion
DEV float fast_tanh(float x) {
  float ax = fabsf(x);
  float e  = __expf(-2.f * ax);
  float r  = (1.f - e) * __builtin_amdgcn_rcpf(1.f + e);
  return copysignf(r, x);
}

// ---------------------------------------------------------------------------
// Pack W (N x K, row-major f32, used as X @ W^T) into WMMA B fragments:
// out[((nt*KT+kt)*32 + lane)*16 + e] = (f16) W[n, k]
// ---------------------------------------------------------------------------
__global__ void packB_kernel(const float* __restrict__ W, _Float16* __restrict__ out,
                             int N, int K) {
  int idx = blockIdx.x * 256 + threadIdx.x;
  int total = N * K;
  if (idx >= total) return;
  int e    = idx & 15;
  int lane = (idx >> 4) & 31;
  int frag = idx >> 9;
  int KT = K >> 5;
  int kt = frag % KT;
  int nt = frag / KT;
  int n  = nt * 16 + (lane & 15);
  int v2 = e >> 1, pp = e & 1;
  int k  = kt * 32 + kpair_of(lane, v2) + pp;
  out[idx] = (_Float16)W[(size_t)n * K + k];
}

// ---------------------------------------------------------------------------
// Sequential scan: RK4 neural ODE + GRU.  One workgroup, 512 threads (16 waves)
// ---------------------------------------------------------------------------
DEV void ode_eval(const _Float16* y16, _Float16* t16, float* stage,
                  const _Float16* w1p, const float* b1,
                  const _Float16* w2p, const float* b2, int wid, int lane) {
  int m8  = (lane >> 4) << 3;
  int col = wid * 16 + (lane & 15);
  v8f c = vzero8();
  mm_tile(y16, HID, w1p, HID / 32, wid, lane, c);
#pragma unroll
  for (int j = 0; j < 8; ++j)
    t16[(j + m8) * HID + col] = (_Float16)fast_tanh(c[j] + b1[col]);
  __syncthreads();
  v8f c2 = vzero8();
  mm_tile(t16, HID, w2p, HID / 32, wid, lane, c2);
#pragma unroll
  for (int j = 0; j < 8; ++j)
    stage[(j + m8) * HID + col] = c2[j] + b2[col];
  __syncthreads();
}

__global__ __launch_bounds__(512) void scan_kernel(
    const float* __restrict__ x, const float* __restrict__ ts,
    const _Float16* __restrict__ w1p, const float* __restrict__ b1,
    const _Float16* __restrict__ w2p, const float* __restrict__ b2,
    const _Float16* __restrict__ wihp, const _Float16* __restrict__ whhp,
    const float* __restrict__ bih, const float* __restrict__ bhh,
    float* __restrict__ Hout) {
  __shared__ float    h_s[Bsz * HID];      // 16 KB
  __shared__ float    stage[Bsz * HID];    // 16 KB
  __shared__ float    fsum[Bsz * HID];     // 16 KB
  __shared__ _Float16 y16[Bsz * HID];      // 8 KB
  __shared__ _Float16 t16[Bsz * HID];      // 8 KB
  __shared__ float    gi[Bsz * 3 * HID];   // 48 KB
  __shared__ float    gh[Bsz * 3 * HID];   // 48 KB
  __shared__ _Float16 x16[Bsz * INF_];     // 2 KB
  __shared__ float    dt_s[Bsz];

  const int tid = threadIdx.x, wid = tid >> 5, lane = tid & 31;
  const int m8  = (lane >> 4) << 3;

  for (int i = tid; i < Bsz * HID; i += 512) h_s[i] = 0.f;
  __syncthreads();

  for (int t = 0; t < Tlen; ++t) {
    if (tid < Bsz)
      dt_s[tid] = (t == 0) ? 0.f : (ts[tid * Tlen + t] - ts[tid * Tlen + t - 1]);
    for (int i = tid; i < Bsz * HID; i += 512) y16[i] = (_Float16)h_s[i];
    __syncthreads();

    // f1
    ode_eval(y16, t16, stage, w1p, b1, w2p, b2, wid, lane);
    for (int i = tid; i < Bsz * HID; i += 512) {
      float f = stage[i]; fsum[i] = f;
      y16[i] = (_Float16)(h_s[i] + 0.5f * dt_s[i >> 8] * f);
    }
    __syncthreads();
    // f2
    ode_eval(y16, t16, stage, w1p, b1, w2p, b2, wid, lane);
    for (int i = tid; i < Bsz * HID; i += 512) {
      float f = stage[i]; fsum[i] += 2.f * f;
      y16[i] = (_Float16)(h_s[i] + 0.5f * dt_s[i >> 8] * f);
    }
    __syncthreads();
    // f3
    ode_eval(y16, t16, stage, w1p, b1, w2p, b2, wid, lane);
    for (int i = tid; i < Bsz * HID; i += 512) {
      float f = stage[i]; fsum[i] += 2.f * f;
      y16[i] = (_Float16)(h_s[i] + dt_s[i >> 8] * f);
    }
    __syncthreads();
    // f4 + combine
    ode_eval(y16, t16, stage, w1p, b1, w2p, b2, wid, lane);
    for (int i = tid; i < Bsz * HID; i += 512) {
      float hn = h_s[i] + dt_s[i >> 8] * (1.f / 6.f) * (fsum[i] + stage[i]);
      h_s[i] = hn;
      y16[i] = (_Float16)hn;
    }
    for (int i = tid; i < Bsz * INF_; i += 512) {
      int m = i >> 6, k = i & 63;
      x16[i] = (_Float16)x[((size_t)m * Tlen + t) * INF_ + k];
    }
    __syncthreads();

    // GRU gates: 48 n-tiles over 16 waves
#pragma unroll
    for (int it = 0; it < 3; ++it) {
      int nt  = wid + 16 * it;
      int col = nt * 16 + (lane & 15);
      v8f cg = vzero8();
      mm_tile(x16, INF_, wihp, INF_ / 32, nt, lane, cg);
#pragma unroll
      for (int j = 0; j < 8; ++j) gi[(j + m8) * 3 * HID + col] = cg[j] + bih[col];
      v8f ch = vzero8();
      mm_tile(y16, HID, whhp, HID / 32, nt, lane, ch);
#pragma unroll
      for (int j = 0; j < 8; ++j) gh[(j + m8) * 3 * HID + col] = ch[j] + bhh[col];
    }
    __syncthreads();

    for (int i = tid; i < Bsz * HID; i += 512) {
      int m = i >> 8, jj = i & 255;
      float ir = gi[m * 3 * HID + jj],       hr = gh[m * 3 * HID + jj];
      float iz = gi[m * 3 * HID + HID + jj], hz = gh[m * 3 * HID + HID + jj];
      float in_= gi[m * 3 * HID + 2*HID + jj], hn = gh[m * 3 * HID + 2*HID + jj];
      float r = sigmoidf_(ir + hr);
      float z = sigmoidf_(iz + hz);
      float n = fast_tanh(in_ + r * hn);
      float hv = (1.f - z) * n + z * h_s[i];
      h_s[i] = hv;
      Hout[((size_t)m * Tlen + t) * HID + jj] = hv;
    }
    __syncthreads();
  }
}

// ---------------------------------------------------------------------------
// LayerNorm rows (256 wide) -> f16.  8 waves/block, 1 wave per row.
// ---------------------------------------------------------------------------
__global__ __launch_bounds__(256) void ln_kernel(const float* __restrict__ src,
                                                 const float* __restrict__ g,
                                                 const float* __restrict__ b,
                                                 _Float16* __restrict__ dst) {
  int wid = threadIdx.x >> 5, lane = threadIdx.x & 31;
  size_t row = (size_t)blockIdx.x * 8 + wid;
  const float* p = src + row * HID;
  float vals[8], s = 0.f;
#pragma unroll
  for (int j = 0; j < 8; ++j) { vals[j] = p[lane + 32 * j]; s += vals[j]; }
#pragma unroll
  for (int off = 16; off >= 1; off >>= 1) s += __shfl_xor(s, off, 32);
  float mean = s * (1.f / HID);
  float q = 0.f;
#pragma unroll
  for (int j = 0; j < 8; ++j) { float d = vals[j] - mean; q += d * d; }
#pragma unroll
  for (int off = 16; off >= 1; off >>= 1) q += __shfl_xor(q, off, 32);
  float rstd = rsqrtf(q * (1.f / HID) + 1e-5f);
#pragma unroll
  for (int j = 0; j < 8; ++j) {
    int c = lane + 32 * j;
    dst[row * HID + c] = (_Float16)((vals[j] - mean) * rstd * g[c] + b[c]);
  }
}

// final: out = H + LN(resid)
__global__ __launch_bounds__(256) void lnadd_kernel(const float* __restrict__ src,
                                                    const float* __restrict__ Hsrc,
                                                    const float* __restrict__ g,
                                                    const float* __restrict__ b,
                                                    float* __restrict__ out) {
  int wid = threadIdx.x >> 5, lane = threadIdx.x & 31;
  size_t row = (size_t)blockIdx.x * 8 + wid;
  const float* p = src + row * HID;
  float vals[8], s = 0.f;
#pragma unroll
  for (int j = 0; j < 8; ++j) { vals[j] = p[lane + 32 * j]; s += vals[j]; }
#pragma unroll
  for (int off = 16; off >= 1; off >>= 1) s += __shfl_xor(s, off, 32);
  float mean = s * (1.f / HID);
  float q = 0.f;
#pragma unroll
  for (int j = 0; j < 8; ++j) { float d = vals[j] - mean; q += d * d; }
#pragma unroll
  for (int off = 16; off >= 1; off >>= 1) q += __shfl_xor(q, off, 32);
  float rstd = rsqrtf(q * (1.f / HID) + 1e-5f);
#pragma unroll
  for (int j = 0; j < 8; ++j) {
    int c = lane + 32 * j;
    out[row * HID + c] = Hsrc[row * HID + c] +
                         (vals[j] - mean) * rstd * g[c] + b[c];
  }
}

// ---------------------------------------------------------------------------
// GEMM: Y = act(X16 @ Wpacked^T + bias) [+= resid in/out f32] [-> f16 out]
// grid = (ROWS/16, N/64); 128 threads (4 waves), 1 n-tile per wave.
// ---------------------------------------------------------------------------
__global__ __launch_bounds__(128) void gemm_kernel(
    const _Float16* __restrict__ X, const _Float16* __restrict__ Wp,
    const float* __restrict__ bias, float* __restrict__ residIO,
    _Float16* __restrict__ outH, int N, int K, int relu) {
  int wid = threadIdx.x >> 5, lane = threadIdx.x & 31;
  int rt = blockIdx.x;
  int ntile = blockIdx.y * 4 + wid;
  int KT = K >> 5;
  v8f acc = vzero8();
  const _Float16* abase = X + (size_t)rt * 16 * K;
  for (int kt = 0; kt < KT; ++kt) {
    // prefetch next B fragment of the weight stream (global_prefetch_b8)
    if (kt + 1 < KT)
      __builtin_prefetch(Wp + (((size_t)ntile * KT + kt + 1) * 32 + lane) * 16, 0, 0);
    v16h a = load_a_frag(abase + kt * 32, K, lane);
    v16h b = *(const v16h*)(Wp + (((size_t)ntile * KT + kt) * 32 + lane) * 16);
    acc = wmma16(a, b, acc);
  }
  int m8 = (lane >> 4) << 3;
  int col = ntile * 16 + (lane & 15);
#pragma unroll
  for (int j = 0; j < 8; ++j) {
    size_t row = (size_t)rt * 16 + j + m8;
    float v = acc[j] + bias[col];
    if (relu) v = fmaxf(v, 0.f);
    if (residIO) { v += residIO[row * N + col]; residIO[row * N + col] = v; }
    if (outH) outH[row * (size_t)N + col] = (_Float16)v;
  }
}

// ---------------------------------------------------------------------------
// Causal flash attention. grid=(B*HEADS, T/16), 1 wave per block.
// qkv layout: [row=b*T+t][768] = (q|k|v) each HEADS*DH.
// ---------------------------------------------------------------------------
__global__ __launch_bounds__(32) void attn_kernel(const _Float16* __restrict__ qkv,
                                                  _Float16* __restrict__ ctx) {
  __shared__ _Float16 p_lds[16 * 32];
  int lane = threadIdx.x;
  int bh = blockIdx.x, b = bh >> 2, hd = bh & 3;
  int qt = blockIdx.y;
  size_t rowbase = (size_t)b * Tlen;
  const float scale = 0.125f;   // 1/sqrt(64)
  int m8 = (lane >> 4) << 3;

  v16h aq0 = load_a_frag(qkv + (rowbase + qt * 16) * 768 + hd * DH + 0,  768, lane);
  v16h aq1 = load_a_frag(qkv + (rowbase + qt * 16) * 768 + hd * DH + 32, 768, lane);

  v8f acc[4];
#pragma unroll
  for (int nt = 0; nt < 4; ++nt) acc[nt] = vzero8();
  float mrun[8], lrun[8];
#pragma unroll
  for (int j = 0; j < 8; ++j) { mrun[j] = -3e38f; lrun[j] = 0.f; }

  int nk = qt * 16 + 16;              // causal: keys [0, nk)
  int nchunks = (nk + 31) >> 5;
  for (int kc = 0; kc < nchunks; ++kc) {
    int keybase = kc << 5;
    float sarr[2][8];
#pragma unroll
    for (int t2 = 0; t2 < 2; ++t2) {
      int keyt = keybase + t2 * 16;
      if (keyt < nk) {
        v8f s = vzero8();
        v16h bk0 = load_a_frag(qkv + (rowbase + keyt) * 768 + HID + hd * DH + 0,  768, lane);
        v16h bk1 = load_a_frag(qkv + (rowbase + keyt) * 768 + HID + hd * DH + 32, 768, lane);
        s = wmma16(aq0, bk0, s);
        s = wmma16(aq1, bk1, s);
#pragma unroll
        for (int j = 0; j < 8; ++j) {
          int qi = qt * 16 + j + m8;
          int kj = keyt + (lane & 15);
          float sv = s[j] * scale;
          sarr[t2][j] = (kj > qi) ? -3e38f : sv;
        }
      } else {
#pragma unroll
        for (int j = 0; j < 8; ++j) sarr[t2][j] = -3e38f;
      }
    }
    // online softmax update
#pragma unroll
    for (int j = 0; j < 8; ++j) {
      float r = fmaxf(sarr[0][j], sarr[1][j]);
#pragma unroll
      for (int off = 8; off >= 1; off >>= 1) r = fmaxf(r, __shfl_xor(r, off, 32));
      float nm = fmaxf(mrun[j], r);
      float alpha = __expf(mrun[j] - nm);
      mrun[j] = nm;
      float p0 = __expf(sarr[0][j] - nm);
      float p1 = __expf(sarr[1][j] - nm);
      float rs = p0 + p1;
#pragma unroll
      for (int off = 8; off >= 1; off >>= 1) rs += __shfl_xor(rs, off, 32);
      lrun[j] = lrun[j] * alpha + rs;
#pragma unroll
      for (int nt = 0; nt < 4; ++nt) acc[nt][j] *= alpha;
      p_lds[(j + m8) * 32 + (lane & 15)]      = (_Float16)p0;
      p_lds[(j + m8) * 32 + 16 + (lane & 15)] = (_Float16)p1;
    }
    __syncthreads();
    v16h ap = load_a_frag(p_lds, 32, lane);
#pragma unroll
    for (int nt = 0; nt < 4; ++nt) {
      v16h bv = load_b_rows(qkv + (rowbase + keybase) * 768 + 2 * HID + hd * DH + nt * 16,
                            768, lane);
      acc[nt] = wmma16(ap, bv, acc[nt]);
    }
    __syncthreads();
  }
  float rl[8];
#pragma unroll
  for (int j = 0; j < 8; ++j) rl[j] = __builtin_amdgcn_rcpf(lrun[j]);
#pragma unroll
  for (int nt = 0; nt < 4; ++nt)
#pragma unroll
    for (int j = 0; j < 8; ++j) {
      int m = j + m8;
      int col = hd * DH + nt * 16 + (lane & 15);
      ctx[(rowbase + qt * 16 + m) * (size_t)HID + col] =
          (_Float16)(acc[nt][j] * rl[j]);
    }
}

// ---------------------------------------------------------------------------
extern "C" void kernel_launch(void* const* d_in, const int* in_sizes, int n_in,
                              void* d_out, int out_size, void* d_ws, size_t ws_size,
                              hipStream_t stream) {
  const float* x        = (const float*)d_in[0];
  const float* ts       = (const float*)d_in[1];
  const float* ode_w1   = (const float*)d_in[2];
  const float* ode_b1   = (const float*)d_in[3];
  const float* ode_w2   = (const float*)d_in[4];
  const float* ode_b2   = (const float*)d_in[5];
  const float* gru_wih  = (const float*)d_in[6];
  const float* gru_whh  = (const float*)d_in[7];
  const float* gru_bih  = (const float*)d_in[8];
  const float* gru_bhh  = (const float*)d_in[9];
  const float* inproj_w = (const float*)d_in[10];
  const float* inproj_b = (const float*)d_in[11];
  const float* outproj_w= (const float*)d_in[12];
  const float* outproj_b= (const float*)d_in[13];
  const float* ff_w1    = (const float*)d_in[14];
  const float* ff_b1    = (const float*)d_in[15];
  const float* ff_w2    = (const float*)d_in[16];
  const float* ff_b2    = (const float*)d_in[17];
  const float* ln1_g    = (const float*)d_in[18];
  const float* ln1_b    = (const float*)d_in[19];
  const float* ln2_g    = (const float*)d_in[20];
  const float* ln2_b    = (const float*)d_in[21];
  const float* lnf_g    = (const float*)d_in[22];
  const float* lnf_b    = (const float*)d_in[23];

  char* ws = (char*)d_ws;
  size_t off = 0;
  auto alloc = [&](size_t bytes) -> char* {
    char* p = ws + off;
    off += (bytes + 255) & ~(size_t)255;
    return p;
  };
  _Float16* w1p     = (_Float16*)alloc((size_t)HID * HID * 2);
  _Float16* w2p     = (_Float16*)alloc((size_t)HID * HID * 2);
  _Float16* wihp    = (_Float16*)alloc((size_t)3 * HID * INF_ * 2);
  _Float16* whhp    = (_Float16*)alloc((size_t)3 * HID * HID * 2);
  _Float16* inprojp = (_Float16*)alloc((size_t)LAYERS * 3 * HID * HID * 2);
  _Float16* outprojp= (_Float16*)alloc((size_t)LAYERS * HID * HID * 2);
  _Float16* ff1p    = (_Float16*)alloc((size_t)LAYERS * FF * HID * 2);
  _Float16* ff2p    = (_Float16*)alloc((size_t)LAYERS * HID * FF * 2);
  float*    Hbuf    = (float*)alloc((size_t)ROWS * HID * 4);
  float*    resid   = (float*)alloc((size_t)ROWS * HID * 4);
  _Float16* z16     = (_Float16*)alloc((size_t)ROWS * HID * 2);
  _Float16* big16   = (_Float16*)alloc((size_t)ROWS * FF * 2);   // qkv16 / ffh16 union
  _Float16* ctx16   = (_Float16*)alloc((size_t)ROWS * HID * 2);
  _Float16* qkv16 = big16;
  _Float16* ffh16 = big16;

  auto pack = [&](const float* W, _Float16* dst, int N, int K) {
    int total = N * K;
    packB_kernel<<<(total + 255) / 256, 256, 0, stream>>>(W, dst, N, K);
  };
  pack(ode_w1, w1p, HID, HID);
  pack(ode_w2, w2p, HID, HID);
  pack(gru_wih, wihp, 3 * HID, INF_);
  pack(gru_whh, whhp, 3 * HID, HID);
  for (int l = 0; l < LAYERS; ++l) {
    pack(inproj_w  + (size_t)l * 3 * HID * HID, inprojp  + (size_t)l * 3 * HID * HID, 3 * HID, HID);
    pack(outproj_w + (size_t)l * HID * HID,     outprojp + (size_t)l * HID * HID,     HID, HID);
    pack(ff_w1     + (size_t)l * FF * HID,      ff1p     + (size_t)l * FF * HID,      FF, HID);
    pack(ff_w2     + (size_t)l * HID * FF,      ff2p     + (size_t)l * HID * FF,      HID, FF);
  }

  // Phase 1: sequential scan (1 workgroup, 16 waves)
  scan_kernel<<<1, 512, 0, stream>>>(x, ts, w1p, ode_b1, w2p, ode_b2,
                                     wihp, whhp, gru_bih, gru_bhh, Hbuf);

  // residual stream starts as H
  (void)hipMemcpyAsync(resid, Hbuf, (size_t)ROWS * HID * 4,
                       hipMemcpyDeviceToDevice, stream);

  auto gemm = [&](const _Float16* X, const _Float16* Wp, const float* bias,
                  float* residIO, _Float16* outH, int N, int K, int relu) {
    dim3 grid(ROWS / 16, N / 64);
    gemm_kernel<<<grid, 128, 0, stream>>>(X, Wp, bias, residIO, outH, N, K, relu);
  };

  for (int l = 0; l < LAYERS; ++l) {
    // --- attention sublayer ---
    ln_kernel<<<ROWS / 8, 256, 0, stream>>>(resid, ln1_g + l * HID, ln1_b + l * HID, z16);
    gemm(z16, inprojp + (size_t)l * 3 * HID * HID, inproj_b + (size_t)l * 3 * HID,
         nullptr, qkv16, 3 * HID, HID, 0);
    {
      dim3 ag(Bsz * HEADS, Tlen / 16);
      attn_kernel<<<ag, 32, 0, stream>>>(qkv16, ctx16);
    }
    gemm(ctx16, outprojp + (size_t)l * HID * HID, outproj_b + (size_t)l * HID,
         resid, nullptr, HID, HID, 0);
    // --- feed-forward sublayer ---
    ln_kernel<<<ROWS / 8, 256, 0, stream>>>(resid, ln2_g + l * HID, ln2_b + l * HID, z16);
    gemm(z16, ff1p + (size_t)l * FF * HID, ff_b1 + (size_t)l * FF,
         nullptr, ffh16, FF, HID, 1);
    gemm(ffh16, ff2p + (size_t)l * HID * FF, ff_b2 + (size_t)l * HID,
         resid, nullptr, HID, FF, 0);
  }

  lnadd_kernel<<<ROWS / 8, 256, 0, stream>>>(resid, Hbuf, lnf_g, lnf_b, (float*)d_out);
}
